// JointAttention_88124138979477
// MI455X (gfx1250) — matrix operand order repaired and checked by reading, not compiled, not run on hardware
//
#include <hip/hip_runtime.h>
#include <hip/hip_bf16.h>

typedef _Float16 half_t;
typedef __attribute__((ext_vector_type(16))) _Float16 v16h;
typedef __attribute__((ext_vector_type(8)))  _Float16 v8h;
typedef __attribute__((ext_vector_type(4)))  _Float16 v4h;
typedef __attribute__((ext_vector_type(8)))  float    v8f;
typedef __attribute__((ext_vector_type(4)))  float    v4f;
typedef __attribute__((vector_size(16)))     int      v4i_t;

#define DIM   2048
#define NH    16
#define NKV   4
#define HD    128
#define BSZ   2
#define SEQ   2048
#define TOK   (BSZ*SEQ)          // 4096
#define QKV_N ((NH+2*NKV)*HD)    // 3072

// ---- CDNA5 async Global->LDS copy (ASYNCcnt-tracked), with fallback --------
#if defined(__HIP_DEVICE_COMPILE__) && __has_builtin(__builtin_amdgcn_global_load_async_to_lds_b128)
#define USE_ASYNC 1
#else
#define USE_ASYNC 0
#endif

#if USE_ASYNC
#if __has_builtin(__builtin_amdgcn_s_wait_asynccnt)
#define WAIT_ASYNC() __builtin_amdgcn_s_wait_asynccnt(0)
#else
#define WAIT_ASYNC() asm volatile("s_wait_asynccnt 0x0" ::: "memory")
#endif
__device__ __forceinline__ void async_copy_b128(const void* g, void* l) {
  __builtin_amdgcn_global_load_async_to_lds_b128(
      (__attribute__((address_space(1))) v4i_t*)g,
      (__attribute__((address_space(3))) v4i_t*)l, 0, 0);
}
#endif

// ---------------------------------------------------------------------------
// 1) Row-wise L2 normalization of a weight matrix [rows, K], output fp16.
// ---------------------------------------------------------------------------
__global__ __launch_bounds__(256)
void k_rownorm(const float* __restrict__ w, half_t* __restrict__ out, int K) {
  __shared__ float red[256];
  const int r = blockIdx.x;
  const float* row = w + (size_t)r * K;
  float s = 0.f;
  for (int c = threadIdx.x; c < K; c += 256) { float v = row[c]; s += v * v; }
  red[threadIdx.x] = s;
  __syncthreads();
  for (int st = 128; st > 0; st >>= 1) {
    if (threadIdx.x < st) red[threadIdx.x] += red[threadIdx.x + st];
    __syncthreads();
  }
  const float inv = 1.0f / (sqrtf(red[0]) + 1e-6f);
  for (int c = threadIdx.x; c < K; c += 256)
    out[(size_t)r * K + c] = (half_t)(row[c] * inv);
}

// ---------------------------------------------------------------------------
// 2) fp32 -> fp16 elementwise (vectorized x4)
// ---------------------------------------------------------------------------
__global__ __launch_bounds__(256)
void k_f32_to_f16(const float* __restrict__ x, half_t* __restrict__ o, int n4) {
  int i = blockIdx.x * blockDim.x + threadIdx.x;
  if (i < n4) {
    v4f v = ((const v4f*)x)[i];
    v4h h; h[0] = (half_t)v[0]; h[1] = (half_t)v[1];
           h[2] = (half_t)v[2]; h[3] = (half_t)v[3];
    ((v4h*)o)[i] = h;
  }
}

// ---------------------------------------------------------------------------
// 3) Tiled WMMA GEMM: C[M,N] = A[M,K] * Bw[N,K]^T   (A,Bw fp16, C fp32)
//    Block tile 64(M) x 128(N), 8 waves of 32x32, K-step 32.
//    Double-buffered LDS staged with async Global->LDS copies.
// ---------------------------------------------------------------------------
__global__ __launch_bounds__(256)
void k_gemm_wmma(const half_t* __restrict__ A, const half_t* __restrict__ Bw,
                 float* __restrict__ C, int M, int N, int K) {
  __shared__ half_t As[2][64][48];    // 48-half stride keeps 16B alignment
  __shared__ half_t Bs[2][128][48];
  const int tid  = threadIdx.x;
  const int lane = tid & 31, w = tid >> 5;
  const int la = lane & 15, lh = lane >> 4;
  const int n0 = blockIdx.x * 128, m0 = blockIdx.y * 64;
  const int wm = (w >> 2) * 32, wn = (w & 3) * 32;
  v8f acc[2][2] = {};

  const int arow = tid >> 2, ac8 = (tid & 3) * 8;   // A tile: 64 rows x 32 halves
  const int brow = tid >> 1, bc16 = (tid & 1) * 16; // B tile: 128 rows x 32 halves
  const half_t* gA = &A[(size_t)(m0 + arow) * K + ac8];
  const half_t* gB = &Bw[(size_t)(n0 + brow) * K + bc16];

  auto stage = [&](int buf, int k0) {
#if USE_ASYNC
    async_copy_b128(gA + k0,     &As[buf][arow][ac8]);
    async_copy_b128(gB + k0,     &Bs[buf][brow][bc16]);
    async_copy_b128(gB + k0 + 8, &Bs[buf][brow][bc16 + 8]);
#else
    *(v8h*)&As[buf][arow][ac8]      = *(const v8h*)(gA + k0);
    *(v8h*)&Bs[buf][brow][bc16]     = *(const v8h*)(gB + k0);
    *(v8h*)&Bs[buf][brow][bc16 + 8] = *(const v8h*)(gB + k0 + 8);
#endif
  };

  stage(0, 0);
  int buf = 0;
  for (int k0 = 0; k0 < K; k0 += 32, buf ^= 1) {
#if USE_ASYNC
    WAIT_ASYNC();                 // my async writes into `buf` are done
#endif
    __syncthreads();              // everyone's writes visible / prior reads done
    if (k0 + 32 < K) stage(buf ^ 1, k0 + 32);   // overlap copy with compute
    if (k0 + 64 < K) {
      __builtin_prefetch(gA + k0 + 64, 0, 3);
      __builtin_prefetch(gB + k0 + 64, 0, 3);
    }

    v16h af[2], bf[2];
#pragma unroll
    for (int am = 0; am < 2; ++am) {
      const int m = wm + am * 16 + la;               // A: lane = M row
      v8h lo = *(const v8h*)&As[buf][m][lh * 8];     // K = lh*8 .. +7
      v8h hi = *(const v8h*)&As[buf][m][lh * 8 + 16];
#pragma unroll
      for (int j = 0; j < 8; ++j) { af[am][j] = lo[j]; af[am][8 + j] = hi[j]; }
    }
#pragma unroll
    for (int bn = 0; bn < 2; ++bn) {
      const int n = wn + bn * 16 + la;               // B: lane = N column
      v8h lo = *(const v8h*)&Bs[buf][n][lh * 8];     // K = lh*8 .. +7
      v8h hi = *(const v8h*)&Bs[buf][n][lh * 8 + 16];
#pragma unroll
      for (int j = 0; j < 8; ++j) { bf[bn][j] = lo[j]; bf[bn][8 + j] = hi[j]; }
    }
#pragma unroll
    for (int am = 0; am < 2; ++am)
#pragma unroll
      for (int bn = 0; bn < 2; ++bn)
        acc[am][bn] = __builtin_amdgcn_wmma_f32_16x16x32_f16(
            false, af[am], false, bf[bn], (short)0, acc[am][bn], false, false);
  }

#pragma unroll
  for (int am = 0; am < 2; ++am)
#pragma unroll
    for (int bn = 0; bn < 2; ++bn)
#pragma unroll
      for (int r = 0; r < 8; ++r) {
        const int row = m0 + wm + am * 16 + r + lh * 8;  // C: VGPR r -> M=r(+8)
        const int col = n0 + wn + bn * 16 + la;          //    lane   -> N
        C[(size_t)row * N + col] = acc[am][bn][r];
      }
}

// ---------------------------------------------------------------------------
// 4) RoPE + per-head L2 norm (q,k) + scale, fp16 outputs.
//    V is stored transposed [b, kvh, d, s] for the PV WMMA B-fragment.
//    Grid: (TOK, 3) x 256; wave -> one of 24 head-slots per token.
// ---------------------------------------------------------------------------
__global__ __launch_bounds__(256)
void k_rope_norm(const float* __restrict__ qkv, const float* __restrict__ fcos,
                 const float* __restrict__ fsin, const float* __restrict__ s_q,
                 const float* __restrict__ s_k, half_t* __restrict__ qh,
                 half_t* __restrict__ kh, half_t* __restrict__ vt) {
  const int t = blockIdx.x;               // token 0..4095
  const int b = t >> 11, s = t & 2047;
  const int lane = threadIdx.x & 31, w = threadIdx.x >> 5;
  const int slot = blockIdx.y * 8 + w;    // 0..23
  const float* row = qkv + (size_t)t * QKV_N;

  int colbase, kind, hidx;
  if (slot < 16)      { kind = 0; hidx = slot;      colbase = hidx * HD; }
  else if (slot < 20) { kind = 1; hidx = slot - 16; colbase = 2048 + hidx * HD; }
  else                { kind = 2; hidx = slot - 20; colbase = 2560 + hidx * HD; }

  v4f v = *(const v4f*)&row[colbase + lane * 4];

  if (kind == 2) {  // V: fp16 transpose-store vt[((b*NKV+kvh)*HD + d)*SEQ + s]
    const size_t base = ((size_t)(b * NKV + hidx) * HD) * SEQ + s;
#pragma unroll
    for (int e = 0; e < 4; ++e)
      vt[base + (size_t)(lane * 4 + e) * SEQ] = (half_t)v[e];
    return;
  }
  // RoPE: consecutive pairs; pair indices i0 = 2*lane, i1 = 2*lane+1
  const int i0 = lane * 2, i1 = lane * 2 + 1;
  const float c0 = fcos[s * 64 + i0], sn0 = fsin[s * 64 + i0];
  const float c1 = fcos[s * 64 + i1], sn1 = fsin[s * 64 + i1];
  const float r0  = v[0] * c0 - v[1] * sn0;
  const float im0 = v[0] * sn0 + v[1] * c0;
  const float r1  = v[2] * c1 - v[3] * sn1;
  const float im1 = v[2] * sn1 + v[3] * c1;

  float ss = r0 * r0 + im0 * im0 + r1 * r1 + im1 * im1;
#pragma unroll
  for (int m = 16; m >= 1; m >>= 1) ss += __shfl_xor(ss, m, 32);
  const float inv = 1.0f / (sqrtf(ss) + 1e-12f);
  const float* sc = (kind == 0) ? s_q : s_k;
  v4h h;
  h[0] = (half_t)(r0  * inv * sc[lane * 4 + 0]);
  h[1] = (half_t)(im0 * inv * sc[lane * 4 + 1]);
  h[2] = (half_t)(r1  * inv * sc[lane * 4 + 2]);
  h[3] = (half_t)(im1 * inv * sc[lane * 4 + 3]);
  if (kind == 0)
    *(v4h*)&qh[((size_t)t * NH + hidx) * HD + lane * 4] = h;
  else
    *(v4h*)&kh[(((size_t)(b * NKV + hidx)) * SEQ + s) * HD + lane * 4] = h;
}

// ---------------------------------------------------------------------------
// 5) Flash-style attention. Block = 8 waves sharing one (b, head); K/V tiles
//    for each 32-key step cooperatively async-staged into LDS; each wave owns
//    a 16-row q tile with online softmax.
//    Grid (SEQ/128, NH, BSZ) x 256.
// ---------------------------------------------------------------------------
__global__ __launch_bounds__(256)
void k_attention(const half_t* __restrict__ qh, const half_t* __restrict__ kh,
                 const half_t* __restrict__ vt, const int* __restrict__ mask,
                 half_t* __restrict__ attn) {
  __shared__ half_t Ks[32][128];     // [key][d]   8 KB
  __shared__ half_t Vs[128][32];     // [d][key]   8 KB
  __shared__ half_t Ps[8][16][32];   // per-wave P staging  8 KB
  const int tid = threadIdx.x;
  const int lane = tid & 31, w = tid >> 5;
  const int la = lane & 15, lh = lane >> 4;
  const int h = blockIdx.y, b = blockIdx.z;
  const int kvh = h >> 2;                     // GQA: 4 q heads per kv head
  const int qbase = blockIdx.x * 128 + w * 16;
  const float SCALE = 11.313708498984761f;    // sqrt(HEAD_DIM) per reference

  // Load 4 Q A-fragments (d in chunks of 32), rows = qbase..qbase+15
  v16h qf[4];
  {
    const half_t* qp = qh + ((size_t)(b * SEQ + qbase + la) * NH + h) * HD;
#pragma unroll
    for (int c = 0; c < 4; ++c) {
      v8h lo = *(const v8h*)&qp[c * 32 + lh * 8];
      v8h hi = *(const v8h*)&qp[c * 32 + lh * 8 + 16];
#pragma unroll
      for (int j = 0; j < 8; ++j) { qf[c][j] = lo[j]; qf[c][8 + j] = hi[j]; }
    }
  }
  const half_t* kbasep = kh + ((size_t)(b * NKV + kvh)) * SEQ * HD;
  const half_t* vbasep = vt + ((size_t)(b * NKV + kvh)) * HD * SEQ;

  // cooperative staging coordinates
  const int krow = tid >> 3, kcol = (tid & 7) * 16;   // 32 rows x 128 halves
  const int vrow = tid >> 1, vcol = (tid & 1) * 16;   // 128 rows x 32 halves
  const half_t* gK = kbasep + (size_t)krow * HD + kcol;
  const half_t* gV = vbasep + (size_t)vrow * SEQ + vcol;

  float m8[8], l8[8];
  v8f o[8] = {};
#pragma unroll
  for (int r = 0; r < 8; ++r) { m8[r] = -1e30f; l8[r] = 0.f; }

  for (int kt = 0; kt < SEQ; kt += 32) {
    __syncthreads();      // previous-iteration consumers of Ks/Vs are done
#if USE_ASYNC
    async_copy_b128(gK + (size_t)kt * HD,     &Ks[krow][kcol]);
    async_copy_b128(gK + (size_t)kt * HD + 8, &Ks[krow][kcol + 8]);
    async_copy_b128(gV + kt,                  &Vs[vrow][vcol]);
    async_copy_b128(gV + kt + 8,              &Vs[vrow][vcol + 8]);
    WAIT_ASYNC();
#else
    *(v8h*)&Ks[krow][kcol]     = *(const v8h*)(gK + (size_t)kt * HD);
    *(v8h*)&Ks[krow][kcol + 8] = *(const v8h*)(gK + (size_t)kt * HD + 8);
    *(v8h*)&Vs[vrow][vcol]     = *(const v8h*)(gV + kt);
    *(v8h*)&Vs[vrow][vcol + 8] = *(const v8h*)(gV + kt + 8);
#endif
    __syncthreads();      // tile visible to all waves

    // --- scores: two 16x16 tiles over 32 keys, K-dim = head_dim (4x32) ---
    v8f sc0 = {}, sc1 = {};
#pragma unroll
    for (int c = 0; c < 4; ++c) {
      v8h l0 = *(const v8h*)&Ks[la][c * 32 + lh * 8];
      v8h h0 = *(const v8h*)&Ks[la][c * 32 + lh * 8 + 16];
      v8h l1 = *(const v8h*)&Ks[16 + la][c * 32 + lh * 8];
      v8h h1 = *(const v8h*)&Ks[16 + la][c * 32 + lh * 8 + 16];
      v16h kf0, kf1;
#pragma unroll
      for (int j = 0; j < 8; ++j) {
        kf0[j] = l0[j]; kf0[8 + j] = h0[j];
        kf1[j] = l1[j]; kf1[8 + j] = h1[j];
      }
      sc0 = __builtin_amdgcn_wmma_f32_16x16x32_f16(false, qf[c], false, kf0,
                                                   (short)0, sc0, false, false);
      sc1 = __builtin_amdgcn_wmma_f32_16x16x32_f16(false, qf[c], false, kf1,
                                                   (short)0, sc1, false, false);
    }
    const bool ok0 = mask[b * SEQ + kt + la] != 0;
    const bool ok1 = mask[b * SEQ + kt + 16 + la] != 0;
#pragma unroll
    for (int r = 0; r < 8; ++r) {
      sc0[r] = ok0 ? sc0[r] * SCALE : -1e30f;
      sc1[r] = ok1 ? sc1[r] * SCALE : -1e30f;
    }
    // --- online softmax per row (rows live across 16-lane halves) ---
#pragma unroll
    for (int r = 0; r < 8; ++r) {
      float rmax = fmaxf(sc0[r], sc1[r]);
#pragma unroll
      for (int mx = 8; mx >= 1; mx >>= 1) rmax = fmaxf(rmax, __shfl_xor(rmax, mx, 32));
      const float mn = fmaxf(m8[r], rmax);
      const float corr = __expf(m8[r] - mn);
      const float p0 = __expf(sc0[r] - mn);
      const float p1 = __expf(sc1[r] - mn);
      float rs = p0 + p1;
#pragma unroll
      for (int mx = 8; mx >= 1; mx >>= 1) rs += __shfl_xor(rs, mx, 32);
      l8[r] = l8[r] * corr + rs;
      m8[r] = mn;
      sc0[r] = p0; sc1[r] = p1;
#pragma unroll
      for (int dt = 0; dt < 8; ++dt) o[dt][r] *= corr;
    }
    // --- C-layout probs -> A-layout fp16 via per-wave LDS round-trip ---
#pragma unroll
    for (int r = 0; r < 8; ++r) {
      Ps[w][r + lh * 8][la]      = (half_t)sc0[r];
      Ps[w][r + lh * 8][16 + la] = (half_t)sc1[r];
    }
    v16h pf;
    {
      v8h lo = *(const v8h*)&Ps[w][la][lh * 8];
      v8h hi = *(const v8h*)&Ps[w][la][lh * 8 + 16];
#pragma unroll
      for (int j = 0; j < 8; ++j) { pf[j] = lo[j]; pf[8 + j] = hi[j]; }
    }
    // --- PV: out[16 x 128] += P(16x32) * V(32 x 128) from LDS ---
#pragma unroll
    for (int dt = 0; dt < 8; ++dt) {
      v8h lo = *(const v8h*)&Vs[dt * 16 + la][lh * 8];
      v8h hi = *(const v8h*)&Vs[dt * 16 + la][lh * 8 + 16];
      v16h vf;
#pragma unroll
      for (int j = 0; j < 8; ++j) { vf[j] = lo[j]; vf[8 + j] = hi[j]; }
      o[dt] = __builtin_amdgcn_wmma_f32_16x16x32_f16(false, pf, false, vf,
                                                     (short)0, o[dt], false, false);
    }
  }
  // --- epilogue: divide by l, store fp16 attn [tok][h*128+d] ---
#pragma unroll
  for (int r = 0; r < 8; ++r) {
    const float invl = 1.0f / l8[r];
    const int qrow = qbase + r + lh * 8;
    half_t* op = attn + ((size_t)(b * SEQ + qrow)) * DIM + h * HD;
#pragma unroll
    for (int dt = 0; dt < 8; ++dt)
      op[dt * 16 + la] = (half_t)(o[dt][r] * invl);
  }
}

// ---------------------------------------------------------------------------
extern "C" void kernel_launch(void* const* d_in, const int* in_sizes, int n_in,
                              void* d_out, int out_size, void* d_ws, size_t ws_size,
                              hipStream_t stream) {
  (void)in_sizes; (void)n_in; (void)out_size; (void)ws_size;
  const float* x     = (const float*)d_in[0];
  const int*   xmask = (const int*)d_in[1];
  const float* fcos  = (const float*)d_in[2];
  const float* fsin  = (const float*)d_in[3];
  const float* qkv_w = (const float*)d_in[4];
  const float* out_w = (const float*)d_in[5];
  const float* s_q   = (const float*)d_in[6];
  const float* s_k   = (const float*)d_in[7];
  float* out = (float*)d_out;

  char* ws = (char*)d_ws;
  size_t off = 0;
  auto alloc = [&](size_t bytes) -> char* {
    char* p = ws + off;
    off = (off + bytes + 255) & ~(size_t)255;
    return p;
  };
  half_t* wqkv_h = (half_t*)alloc((size_t)QKV_N * DIM * 2);
  half_t* wout_h = (half_t*)alloc((size_t)DIM * DIM * 2);
  half_t* xh     = (half_t*)alloc((size_t)TOK * DIM * 2);
  float*  qkvf   = (float*) alloc((size_t)TOK * QKV_N * 4);
  half_t* qh     = (half_t*)alloc((size_t)TOK * NH * HD * 2);
  half_t* kh     = (half_t*)alloc((size_t)BSZ * NKV * SEQ * HD * 2);
  half_t* vt     = (half_t*)alloc((size_t)BSZ * NKV * HD * SEQ * 2);
  half_t* attnh  = (half_t*)alloc((size_t)TOK * DIM * 2);

  k_rownorm<<<QKV_N, 256, 0, stream>>>(qkv_w, wqkv_h, DIM);
  k_rownorm<<<DIM, 256, 0, stream>>>(out_w, wout_h, DIM);
  const int n4 = TOK * DIM / 4;
  k_f32_to_f16<<<(n4 + 255) / 256, 256, 0, stream>>>(x, xh, n4);
  k_gemm_wmma<<<dim3(QKV_N / 128, TOK / 64), 256, 0, stream>>>(
      xh, wqkv_h, qkvf, TOK, QKV_N, DIM);
  k_rope_norm<<<dim3(TOK, 3), 256, 0, stream>>>(qkvf, fcos, fsin, s_q, s_k,
                                                qh, kh, vt);
  k_attention<<<dim3(SEQ / 128, NH, BSZ), 256, 0, stream>>>(qh, kh, vt, xmask,
                                                            attnh);
  k_gemm_wmma<<<dim3(DIM / 128, TOK / 64), 256, 0, stream>>>(
      attnh, wout_h, out, TOK, DIM, DIM);
}
